// CGRNN_batch_IGraph_33741263078248
// MI455X (gfx1250) — compile-verified, matrix operand order checked by reading, not run on hardware
//
#include <hip/hip_runtime.h>
#include <hip/hip_bf16.h>

// CDNA5 (gfx1250) wave32 WMMA types
typedef __attribute__((ext_vector_type(16))) __bf16 v16bf;
typedef __attribute__((ext_vector_type(8)))  float  v8f;

__device__ __forceinline__ v8f wmma_bf16(v16bf a, v16bf b, v8f c) {
    // D(f32 16x16) = A(16x32 bf16) * B(32x16 bf16) + C
    return __builtin_amdgcn_wmma_f32_16x16x32_bf16(false, a, false, b, (short)0, c,
                                                   false, false);
}

// --- CDNA5 async Global->LDS copy (tracked by ASYNCcnt) ---------------------
__device__ __forceinline__ unsigned lds_off_u32(const void* p) {
    // Generic LDS pointer: addr[31:0] is the wave-relative LDS offset (ISA 10.2)
    return (unsigned)(unsigned long long)(uintptr_t)p;
}
__device__ __forceinline__ void async_copy_b128(unsigned lds_dst,
                                                unsigned long long gsrc) {
    asm volatile("global_load_async_to_lds_b128 %0, %1, off"
                 :: "v"(lds_dst), "v"(gsrc) : "memory");
}
__device__ __forceinline__ void wait_async0() {
    asm volatile("s_wait_asynccnt 0x0" ::: "memory");
}

#define BB 32      // batch (fixed by problem)
#define DD 128     // hidden dim (fixed by problem)
#define FRAG_B 1024            // one 32x16 bf16 B-fragment, lane-major
#define GATE_B (64 * FRAG_B)   // 8 ktiles x 8 ntiles per gate
#define NODE_B (3 * GATE_B)    // r,u,c

// ---------------------------------------------------------------------------
// Kernel 1a: vv[n,e] = relu(vpe @ W1 + b1) @ W2 + b2   (tiny, once)
// ---------------------------------------------------------------------------
extern "C" __global__ void cgrnn_vv(const float* __restrict__ vpe,
                                    const float* __restrict__ W1,
                                    const float* __restrict__ b1,
                                    const float* __restrict__ W2,
                                    const float* __restrict__ b2,
                                    float* __restrict__ vv,
                                    int N, int P, int H, int E) {
    __shared__ float hb[512];
    const int n = blockIdx.x;
    const int j = threadIdx.x;            // H threads
    float s = b1[j];
    for (int p = 0; p < P; ++p) s += vpe[(size_t)n * P + p] * W1[(size_t)p * H + j];
    hb[j] = fmaxf(s, 0.f);
    __syncthreads();
    if (j < E) {
        float o = b2[j];
        for (int q = 0; q < H; ++q) o += hb[q] * W2[(size_t)q * E + j];
        vv[n * E + j] = o;
    }
}

// ---------------------------------------------------------------------------
// Kernel 1b: var_total[b,n] = sum_t mask[b,t,n]
// ---------------------------------------------------------------------------
extern "C" __global__ void cgrnn_vt(const int* __restrict__ mask,
                                    float* __restrict__ vt,
                                    int B, int N, int T) {
    int idx = blockIdx.x * blockDim.x + threadIdx.x;
    if (idx >= B * N) return;
    int b = idx / N, nn = idx % N;
    int s = 0;
    for (int t = 0; t < T; ++t) s += mask[((size_t)b * T + t) * N + nn];
    vt[idx] = (float)s;
}

// ---------------------------------------------------------------------------
// Kernel 2: build W_eff[n] = sum_e vv[n,e]*W[e] directly in WMMA B-fragment
// layout (bf16). Fragment (g,n,kt,nt): lane-major, 16 halves per lane matching
// the 16-bit B operand of v_wmma_f32_16x16x32_bf16.
//   lane<16 : col=lane,    halves 0..7 -> K = kt*32 + 0..7,  8..15 -> K = +16..23
//   lane>=16: col=lane-16, halves 0..7 -> K = kt*32 + 8..15, 8..15 -> K = +24..31
// K axis: 0..127 = obs rows (I-index k), 128..255 = h rows (I-index k+1).
// I-row 128 (rarity) is handled separately as a rank-1 bias (kernel 3).
// ---------------------------------------------------------------------------
extern "C" __global__ void cgrnn_weff(const float* __restrict__ Wr,
                                      const float* __restrict__ Wu,
                                      const float* __restrict__ Wc,
                                      const float* __restrict__ vv,
                                      unsigned char* __restrict__ wfrag,
                                      int N, int E, int I2) {
    const int lane = threadIdx.x;                  // 32 threads
    const int kt = blockIdx.x >> 3, nt = blockIdx.x & 7;
    const int n = blockIdx.y, g = blockIdx.z;
    const float* Wg = (g == 0) ? Wr : (g == 1) ? Wu : Wc;
    float v[8];
    for (int e = 0; e < E && e < 8; ++e) v[e] = vv[n * E + e];
    const int dcol = nt * 16 + (lane & 15);
    const int kadd = (lane & 16) ? 8 : 0;
    __bf16* dst = (__bf16*)(wfrag + (size_t)n * NODE_B + (size_t)g * GATE_B +
                            (size_t)(kt * 8 + nt) * FRAG_B) + lane * 16;
#pragma unroll
    for (int j = 0; j < 16; ++j) {
        int k = kt * 32 + kadd + (j < 8 ? j : 16 + (j - 8));
        int ksrc = (k < DD) ? k : k + 1;           // skip rarity row (I-index 128)
        float s = 0.f;
        for (int e = 0; e < E && e < 8; ++e)
            s += v[e] * Wg[((size_t)e * I2 + ksrc) * DD + dcol];
        dst[j] = (__bf16)s;
    }
}

// ---------------------------------------------------------------------------
// Kernel 3: rank-1 rarity weight and effective bias per (gate, node)
// ---------------------------------------------------------------------------
extern "C" __global__ void cgrnn_bias(const float* __restrict__ Wr,
                                      const float* __restrict__ Wu,
                                      const float* __restrict__ Wc,
                                      const float* __restrict__ br,
                                      const float* __restrict__ bu,
                                      const float* __restrict__ bc,
                                      const float* __restrict__ vv,
                                      float* __restrict__ wrar,
                                      float* __restrict__ beff,
                                      int N, int E, int I2) {
    int n = blockIdx.x, g = blockIdx.y, dcol = threadIdx.x;  // 128 threads
    const float* Wg = (g == 0) ? Wr : (g == 1) ? Wu : Wc;
    const float* bg = (g == 0) ? br : (g == 1) ? bu : bc;
    float sw = 0.f, sb = 0.f;
    for (int e = 0; e < E; ++e) {
        float ve = vv[n * E + e];
        sw += ve * Wg[((size_t)e * I2 + DD) * DD + dcol];  // I-row 128 = rarity
        sb += ve * bg[e * DD + dcol];
    }
    wrar[((size_t)g * N + n) * DD + dcol] = sw;
    beff[((size_t)g * N + n) * DD + dcol] = sb;
}

// ---------------------------------------------------------------------------
// Kernel 4: the recurrence. One workgroup (16 waves) per node; W_eff (192 KB
// bf16) LDS-resident (staged via async Global->LDS); obs[t] double-buffered in
// LDS via global_load_async_to_lds_b128 issued one step ahead; 24 bf16 WMMAs
// per wave per step; rarity folded in as a rank-1 accumulator init.
// ---------------------------------------------------------------------------
extern "C" __global__ __launch_bounds__(512, 1)
void cgrnn_main(const float* __restrict__ obs,          // [B,T,N,D]
                const int* __restrict__ mask,           // [B,T,N]
                const int* __restrict__ lengths,        // [B,1]
                const float* __restrict__ avg,          // [B,T,N]
                const unsigned char* __restrict__ wfrag_g,  // NODE_B bytes/node
                const float* __restrict__ wrar_g,       // [3,N,D]
                const float* __restrict__ beff_g,       // [3,N,D]
                const float* __restrict__ vt_g,         // [B,N]
                float* __restrict__ out,                // [B,N,D]
                int T, int N) {
    __shared__ __align__(16) unsigned char Wl[NODE_B];   // 192 KB bf16 fragments
    __shared__ __align__(16) float obsb[2][BB * DD];     // double-buffered obs[t]
    __shared__ float  h_l[BB * DD];                      // f32 hidden state
    __shared__ __align__(16) __bf16 hr_l[BB * DD];       // bf16 h_reset (A operand)
    __shared__ float  rar_l[BB];
    __shared__ int    upd_l[BB];
    __shared__ float  vt_l[BB];
    __shared__ int    end_l[BB];
    __shared__ float  beff_l[3 * DD];
    __shared__ float  wrar_l[3 * DD];

    const int tid  = threadIdx.x;
    const int n    = blockIdx.x;
    const int lane = tid & 31;
    const int w    = tid >> 5;     // 0..15
    const int mt   = w >> 3;       // 0..1 : batch tile
    const int nt   = w & 7;        // 0..7 : output-dim tile

    const unsigned ldsW   = lds_off_u32(&Wl[0]);
    const unsigned ldsO0  = lds_off_u32(&obsb[0][0]);
    const unsigned ldsO1  = lds_off_u32(&obsb[1][0]);
    const unsigned long long obsB = (unsigned long long)(uintptr_t)obs;

    // Prologue: async-stage W_eff fragments (192 KB) + obs[t=0] block (16 KB).
    {
        const unsigned long long wB =
            (unsigned long long)(uintptr_t)(wfrag_g + (size_t)n * NODE_B);
        for (int i = tid; i < NODE_B / 16; i += 512)
            async_copy_b128(ldsW + (unsigned)i * 16u, wB + (unsigned long long)i * 16u);
        for (int i = tid; i < (BB * DD * 4) / 16; i += 512) {
            int b = i >> 5, c = i & 31;                  // 32 chunks of 16B per row
            unsigned long long g =
                obsB + ((((unsigned long long)b * T + 0) * N + n) * DD) * 4ull + c * 16ull;
            async_copy_b128(ldsO0 + (unsigned)i * 16u, g);
        }
    }
    for (int i = tid; i < 3 * DD; i += 512) {
        int g = i / DD, d = i % DD;
        beff_l[i] = beff_g[((size_t)g * N + n) * DD + d];
        wrar_l[i] = wrar_g[((size_t)g * N + n) * DD + d];
    }
    if (tid < BB) {
        vt_l[tid]  = vt_g[(size_t)tid * N + n];
        end_l[tid] = lengths[tid] - 1;
    }
    for (int i = tid; i < BB * DD; i += 512) h_l[i] = 0.f;
    wait_async0();
    __syncthreads();

    const int d    = nt * 16 + (lane & 15);           // output column
    const int m0   = mt * 16 + ((lane & 16) ? 8 : 0); // first row of this lane's D tile
    const int mr   = mt * 16 + (lane & 15);           // A-operand row
    const int kadd = (lane & 16) ? 8 : 0;

    for (int t = 0; t < T; ++t) {
        // Kick off the async HBM->LDS copy of obs[t+1] (overlaps this step).
        if (t + 1 < T) {
            unsigned dst = ((t + 1) & 1) ? ldsO1 : ldsO0;
            for (int i = tid; i < (BB * DD * 4) / 16; i += 512) {
                int b = i >> 5, c = i & 31;
                unsigned long long g = obsB +
                    ((((unsigned long long)b * T + (t + 1)) * N + n) * DD) * 4ull + c * 16ull;
                async_copy_b128(dst + (unsigned)i * 16u, g);
            }
        }
        if (tid < BB) {
            size_t off = ((size_t)tid * T + t) * N + n;
            rar_l[tid] = 0.5f * tanhf(avg[off] / (vt_l[tid] + 1.f));
            upd_l[tid] = (mask[off] > 0) ? 1 : 0;
        }
        if (t + 2 < T) {  // warm L2 one more step ahead
            int pb = tid >> 4, pc = tid & 15;
            __builtin_prefetch(obs + (((size_t)pb * T + (t + 2)) * N + n) * DD + pc * 8, 0, 0);
        }
        __syncthreads();

        // Accumulator init = b_eff[d] + rarity[m] * w_rar[d]  (rank-1 K=257 term)
        v8f accR, accU, accC;
#pragma unroll
        for (int i = 0; i < 8; ++i) {
            float rr = rar_l[m0 + i];
            accR[i] = beff_l[0 * DD + d] + rr * wrar_l[0 * DD + d];
            accU[i] = beff_l[1 * DD + d] + rr * wrar_l[1 * DD + d];
            accC[i] = beff_l[2 * DD + d] + rr * wrar_l[2 * DD + d];
        }

        // Phase A: r,u over K=256 ([obs|h]); obs-half of c.
        const float* orow = &obsb[t & 1][mr * DD];
#pragma unroll
        for (int kt = 0; kt < 8; ++kt) {
            v16bf a;
            if (kt < 4) {
                int klo = kt * 32 + kadd;
#pragma unroll
                for (int j = 0; j < 8; ++j) a[j]     = (__bf16)orow[klo + j];
#pragma unroll
                for (int j = 0; j < 8; ++j) a[8 + j] = (__bf16)orow[klo + 16 + j];
            } else {
                const float* hrow = &h_l[mr * DD + (kt - 4) * 32 + kadd];
#pragma unroll
                for (int j = 0; j < 8; ++j) a[j]     = (__bf16)hrow[j];
#pragma unroll
                for (int j = 0; j < 8; ++j) a[8 + j] = (__bf16)hrow[16 + j];
            }
            const unsigned char* fb = Wl + (size_t)(kt * 8 + nt) * FRAG_B + lane * 32;
            v16bf br = *(const v16bf*)(fb);
            v16bf bu = *(const v16bf*)(fb + GATE_B);
            accR = wmma_bf16(a, br, accR);
            accU = wmma_bf16(a, bu, accU);
            if (kt < 4) {
                v16bf bc = *(const v16bf*)(fb + 2 * GATE_B);
                accC = wmma_bf16(a, bc, accC);
            }
        }

        float hv[8], uu[8], hres[8];
#pragma unroll
        for (int i = 0; i < 8; ++i) {
            float r = 1.f / (1.f + __expf(-accR[i]));
            uu[i]   = 1.f / (1.f + __expf(-accU[i]));
            hv[i]   = h_l[(m0 + i) * DD + d];
            hres[i] = upd_l[m0 + i] ? r * hv[i] : hv[i];
            hr_l[(m0 + i) * DD + d] = (__bf16)hres[i];
        }
        __syncthreads();

        // Phase B: h_reset-half of c.
#pragma unroll
        for (int kt = 4; kt < 8; ++kt) {
            v16bf a;
            const __bf16* hrow = &hr_l[mr * DD + (kt - 4) * 32 + kadd];
#pragma unroll
            for (int j = 0; j < 8; ++j) a[j]     = hrow[j];
#pragma unroll
            for (int j = 0; j < 8; ++j) a[8 + j] = hrow[16 + j];
            const unsigned char* fb =
                Wl + 2 * GATE_B + (size_t)(kt * 8 + nt) * FRAG_B + lane * 32;
            v16bf bc = *(const v16bf*)(fb);
            accC = wmma_bf16(a, bc, accC);
        }

#pragma unroll
        for (int i = 0; i < 8; ++i) {
            float c  = tanhf(accC[i]);
            float hn = upd_l[m0 + i] ? (1.f - uu[i]) * hres[i] + uu[i] * c : hv[i];
            h_l[(m0 + i) * DD + d] = hn;
            if (end_l[m0 + i] == t) out[((size_t)(m0 + i) * N + n) * DD + d] = hn;
        }
        // Make sure next step's obs buffer has landed before anyone reads it.
        wait_async0();
        __syncthreads();
    }
}

// ---------------------------------------------------------------------------
// Host launcher (graph-capture safe: stream-only, scratch from d_ws)
// ---------------------------------------------------------------------------
extern "C" void kernel_launch(void* const* d_in, const int* in_sizes, int n_in,
                              void* d_out, int out_size, void* d_ws, size_t ws_size,
                              hipStream_t stream) {
    (void)n_in; (void)out_size; (void)ws_size;
    const float* obs     = (const float*)d_in[0];
    const int*   mask    = (const int*)d_in[2];
    const int*   lengths = (const int*)d_in[5];
    const float* avg     = (const float*)d_in[6];
    const float* vpe     = (const float*)d_in[7];
    const float* Wr = (const float*)d_in[8];   const float* br = (const float*)d_in[9];
    const float* Wu = (const float*)d_in[10];  const float* bu = (const float*)d_in[11];
    const float* Wc = (const float*)d_in[12];  const float* bc = (const float*)d_in[13];
    const float* W1 = (const float*)d_in[14];  const float* b1 = (const float*)d_in[15];
    const float* W2 = (const float*)d_in[16];  const float* b2 = (const float*)d_in[17];

    int N = 1;                                   // adj is [N,N]
    while ((long long)(N + 1) * (N + 1) <= (long long)in_sizes[1]) ++N;
    const int B  = in_sizes[5];
    const int T  = in_sizes[3] / B;
    const int E  = in_sizes[17];
    const int P  = in_sizes[7] / N;
    const int H  = in_sizes[15];
    const int I2 = 2 * DD + 1;                   // D==128 assumed by tiling

    unsigned char* ws = (unsigned char*)d_ws;
    size_t off = 0;
    auto carve = [&](size_t bytes) {
        size_t p = off;
        off = (off + bytes + 255) & ~(size_t)255;
        return p;
    };
    float* vv            = (float*)(ws + carve((size_t)N * E * 4));
    float* vt            = (float*)(ws + carve((size_t)B * N * 4));
    float* wrar          = (float*)(ws + carve((size_t)3 * N * DD * 4));
    float* beff          = (float*)(ws + carve((size_t)3 * N * DD * 4));
    unsigned char* wfrag = ws + carve((size_t)N * NODE_B);

    hipLaunchKernelGGL(cgrnn_vv,   dim3(N), dim3(H), 0, stream,
                       vpe, W1, b1, W2, b2, vv, N, P, H, E);
    hipLaunchKernelGGL(cgrnn_vt,   dim3((B * N + 255) / 256), dim3(256), 0, stream,
                       mask, vt, B, N, T);
    hipLaunchKernelGGL(cgrnn_weff, dim3(64, N, 3), dim3(32), 0, stream,
                       Wr, Wu, Wc, vv, wfrag, N, E, I2);
    hipLaunchKernelGGL(cgrnn_bias, dim3(N, 3), dim3(128), 0, stream,
                       Wr, Wu, Wc, br, bu, bc, vv, wrar, beff, N, E, I2);
    hipLaunchKernelGGL(cgrnn_main, dim3(N), dim3(512), 0, stream,
                       obs, mask, lengths, avg, wfrag, wrar, beff, vt,
                       (float*)d_out, T, N);
}
